// HeteroGraphSAGE_69020124446814
// MI455X (gfx1250) — matrix operand order repaired and checked by reading, not compiled, not run on hardware
//
#include <hip/hip_runtime.h>
#include <hip/hip_bf16.h>

// ---------------------------------------------------------------------------
// HeteroGraphSAGE forward on MI455X (gfx1250, wave32, WMMA).
// Pipeline: edge-scatter (native global_atomic_add_f32) -> fused
// mean/GEMM/bias/ReLU using v_wmma_f32_16x16x32_bf16 with bf16x3 split
// accumulation (fp32-class accuracy, 3 bf16 WMMAs per 16x16x32 K-step).
// Whole problem is HBM-bound (~7 GB moved vs ~105 GFLOP): scatter uses
// wave-per-edge fully-coalesced float4 gathers + HW f32 atomics; GEMM keeps
// weights L2-resident and streams activations with clause'd b128 loads.
// ---------------------------------------------------------------------------

#define NN      200000     // nodes per type (N_USER == N_ITEM)
#define DIN     128
#define DH      256
#define DOUT    128
#define DEG_PAD 200704     // 200000 padded to a multiple of 256 floats

typedef __attribute__((ext_vector_type(16))) __bf16 bf16x16;
typedef __attribute__((ext_vector_type(8)))  float  floatx8;

// ---------------------------------------------------------------------------
// WMMA helper: D = A(16x32 bf16) x B(32x16 bf16) + C(16x16 f32)
// ---------------------------------------------------------------------------
__device__ __forceinline__ floatx8 wmma_bf16(bf16x16 a, bf16x16 b, floatx8 c) {
  return __builtin_amdgcn_wmma_f32_16x16x32_bf16(
      /*neg_a=*/false, a, /*neg_b=*/false, b,
      /*c_mod=*/(short)0, c, /*reuse_a=*/false, /*reuse_b=*/false);
}

// Split 16 fp32 values into bf16 hi + bf16 residual lo (bf16x3 scheme).
__device__ __forceinline__ void split16(const float* v, bf16x16& hi, bf16x16& lo) {
#pragma unroll
  for (int i = 0; i < 16; ++i) {
    __bf16 h = (__bf16)v[i];
    hi[i] = h;
    lo[i] = (__bf16)(v[i] - (float)h);
  }
}

// A fragment (16x32, MxK): lane l holds row m = m0+(l&15); half = l>>4.
// Its 16 bf16 elements are K = {k0+half*16 .. +15} (contiguous, matching B).
// One contiguous 64B run -> 4x b128.
__device__ __forceinline__ void load_a_frag(const float* __restrict__ A, int lda,
                                            int m, int half, int k0, float scale,
                                            bf16x16& hi, bf16x16& lo) {
  const float* base = A + (size_t)m * lda + k0 + half * 16;
  float4 f0 = ((const float4*)base)[0];
  float4 f1 = ((const float4*)base)[1];
  float4 f2 = ((const float4*)base)[2];
  float4 f3 = ((const float4*)base)[3];
  float v[16] = {f0.x, f0.y, f0.z, f0.w, f1.x, f1.y, f1.z, f1.w,
                 f2.x, f2.y, f2.z, f2.w, f3.x, f3.y, f3.z, f3.w};
#pragma unroll
  for (int i = 0; i < 16; ++i) v[i] *= scale;
  split16(v, hi, lo);
}

// B fragment (32x16, KxN) with B = W^T, W row-major (N x K):
// lane l holds column n (lanes 0-15: K=k0..k0+15, lanes 16-31: K=k0+16..+31),
// i.e. 16 contiguous floats of W[n]. 64B contiguous per lane.
__device__ __forceinline__ void load_b_frag(const float* __restrict__ W, int ldw,
                                            int n, int half, int k0,
                                            bf16x16& hi, bf16x16& lo) {
  const float* base = W + (size_t)n * ldw + k0 + half * 16;
  float4 f0 = ((const float4*)base)[0];
  float4 f1 = ((const float4*)base)[1];
  float4 f2 = ((const float4*)base)[2];
  float4 f3 = ((const float4*)base)[3];
  float v[16] = {f0.x, f0.y, f0.z, f0.w, f1.x, f1.y, f1.z, f1.w,
                 f2.x, f2.y, f2.z, f2.w, f3.x, f3.y, f3.z, f3.w};
  split16(v, hi, lo);
}

// Accumulate one source: C(16x64) += (A * scale) @ W^T over K.
__device__ __forceinline__ void accum_source(const float* __restrict__ A, float scale,
                                             const float* __restrict__ W, int K,
                                             int m, int n0, int lidx, int half,
                                             floatx8& c0, floatx8& c1,
                                             floatx8& c2, floatx8& c3) {
  for (int k0 = 0; k0 < K; k0 += 32) {
    bf16x16 ahi, alo, bhi, blo;
    load_a_frag(A, K, m, half, k0, scale, ahi, alo);

    load_b_frag(W, K, n0 + 0 + lidx, half, k0, bhi, blo);
    c0 = wmma_bf16(ahi, bhi, c0);
    c0 = wmma_bf16(alo, bhi, c0);
    c0 = wmma_bf16(ahi, blo, c0);

    load_b_frag(W, K, n0 + 16 + lidx, half, k0, bhi, blo);
    c1 = wmma_bf16(ahi, bhi, c1);
    c1 = wmma_bf16(alo, bhi, c1);
    c1 = wmma_bf16(ahi, blo, c1);

    load_b_frag(W, K, n0 + 32 + lidx, half, k0, bhi, blo);
    c2 = wmma_bf16(ahi, bhi, c2);
    c2 = wmma_bf16(alo, bhi, c2);
    c2 = wmma_bf16(ahi, blo, c2);

    load_b_frag(W, K, n0 + 48 + lidx, half, k0, bhi, blo);
    c3 = wmma_bf16(ahi, bhi, c3);
    c3 = wmma_bf16(alo, bhi, c3);
    c3 = wmma_bf16(ahi, blo, c3);
  }
}

// ---------------------------------------------------------------------------
// Fused SAGEConv GEMM:  Y = act( (Agg/deg) @ Wl^T + X @ Wr^T + b )
// Wave-per-16x64-tile; 8 waves per block. M,N,K multiples of 16/64/32.
// ---------------------------------------------------------------------------
__global__ void __launch_bounds__(256, 2)
sage_gemm_wmma(const float* __restrict__ Agg, const float* __restrict__ deg,
               const float* __restrict__ X,
               const float* __restrict__ Wl, const float* __restrict__ Wr,
               const float* __restrict__ bias, float* __restrict__ Y,
               int M, int N, int K, int relu) {
  const int lane   = threadIdx.x & 31;
  const int gwave  = (int)((blockIdx.x * blockDim.x + threadIdx.x) >> 5);
  const int ntiles = N >> 6;                 // 64-wide N tiles
  const int m0     = (gwave / ntiles) << 4;
  const int n0     = (gwave % ntiles) << 6;
  if (m0 >= M) return;

  const int half = lane >> 4;
  const int lidx = lane & 15;
  const int m    = m0 + lidx;                // A-fragment row owned by this lane

  const float invd = 1.0f / fmaxf(deg[m], 1.0f);   // mean normalization

  floatx8 c0 = {}, c1 = {}, c2 = {}, c3 = {};
  accum_source(Agg, invd, Wl, K, m, n0, lidx, half, c0, c1, c2, c3);
  accum_source(X,   1.0f, Wr, K, m, n0, lidx, half, c0, c1, c2, c3);

  // Epilogue: C VGPR g <-> row m0 + half*8 + g, col n0 + j*16 + (lane&15).
#pragma unroll
  for (int j = 0; j < 4; ++j) {
    floatx8 c = (j == 0) ? c0 : (j == 1) ? c1 : (j == 2) ? c2 : c3;
    const int n = n0 + j * 16 + lidx;
    const float b = bias[n];
#pragma unroll
    for (int g = 0; g < 8; ++g) {
      const int row = m0 + half * 8 + g;
      float y = c[g] + b;
      if (relu) y = fmaxf(y, 0.0f);
      Y[(size_t)row * N + n] = y;
    }
  }
}

// ---------------------------------------------------------------------------
// Edge scatter: one wave per edge; lane handles one float4 (D=128 -> 1 iter,
// D=256 -> 2 iters). unsafeAtomicAdd forces the HW GLOBAL_ATOMIC_ADD_F32
// (ISA 15.18.3 op 86) instead of a CAS loop; aggregates are coarse-grained
// device memory so the "unsafe" (fine-grained-memory) caveat doesn't apply.
// ---------------------------------------------------------------------------
__device__ __forceinline__ void atom_add(float* p, float v) {
#if defined(__HIP_DEVICE_COMPILE__)
  unsafeAtomicAdd(p, v);
#else
  __hip_atomic_fetch_add(p, v, __ATOMIC_RELAXED, __HIP_MEMORY_SCOPE_AGENT);
#endif
}

__global__ void __launch_bounds__(256, 4)
sage_scatter(const float* __restrict__ xsrc, const int* __restrict__ esrc,
             const int* __restrict__ edst, float* __restrict__ agg,
             float* __restrict__ deg, int nE, int D) {
  const int gw   = (int)((blockIdx.x * blockDim.x + threadIdx.x) >> 5);
  const int lane = threadIdx.x & 31;
  if (gw >= nE) return;
  const int s = esrc[gw];
  const int d = edst[gw];
  const float4* xs = (const float4*)(xsrc + (size_t)s * D);
  float* ag = agg + (size_t)d * D;
  const int nvec = D >> 2;
  for (int c = lane; c < nvec; c += 32) {
    float4 v = xs[c];
    atom_add(ag + 4 * c + 0, v.x);
    atom_add(ag + 4 * c + 1, v.y);
    atom_add(ag + 4 * c + 2, v.z);
    atom_add(ag + 4 * c + 3, v.w);
  }
  if (deg != nullptr && lane == 0) atom_add(deg + d, 1.0f);
}

__global__ void __launch_bounds__(256)
zero_f4(float4* __restrict__ p, long long n4) {
  long long i = (long long)blockIdx.x * blockDim.x + threadIdx.x;
  if (i < n4) p[i] = make_float4(0.f, 0.f, 0.f, 0.f);
}

// ---------------------------------------------------------------------------
extern "C" void kernel_launch(void* const* d_in, const int* in_sizes, int n_in,
                              void* d_out, int out_size, void* d_ws, size_t ws_size,
                              hipStream_t stream) {
  const float* x_user = (const float*)d_in[0];
  const float* x_item = (const float*)d_in[1];
  const int*   e_u2i  = (const int*)d_in[2];   // [2, nE]: row0 src, row1 dst
  const int*   e_i2u  = (const int*)d_in[3];
  const float* Wl1_ui = (const float*)d_in[4];
  const float* Wr1_ui = (const float*)d_in[5];
  const float* b1_ui  = (const float*)d_in[6];
  const float* Wl1_iu = (const float*)d_in[7];
  const float* Wr1_iu = (const float*)d_in[8];
  const float* b1_iu  = (const float*)d_in[9];
  const float* Wl2_ui = (const float*)d_in[10];
  const float* Wr2_ui = (const float*)d_in[11];
  const float* b2_ui  = (const float*)d_in[12];
  const float* Wl2_iu = (const float*)d_in[13];
  const float* Wr2_iu = (const float*)d_in[14];
  const float* b2_iu  = (const float*)d_in[15];

  const int nE = in_sizes[2] / 2;

  // Workspace layout (~616 MB):
  float* degI = (float*)d_ws;                         // item in-degree
  float* degU = degI + DEG_PAD;                       // user in-degree
  float* aggA = degU + DEG_PAD;                       // NN*256 f32, reused 3x
  float* hI   = aggA + (size_t)NN * DH;               // hidden item (NN x 256)
  float* hU   = hI   + (size_t)NN * DH;               // hidden user (NN x 256)

  float* agg1_item = aggA;                            // NN x 128
  float* agg1_user = aggA + (size_t)NN * DIN;         // NN x 128

  float* o_user = (float*)d_out;                      // NN x 128
  float* o_item = o_user + (size_t)NN * DOUT;         // NN x 128

  const int scatterBlocks = (nE + 7) / 8;             // 8 waves (edges) / block
  const long long zero1_n4 = ((long long)2 * DEG_PAD + (long long)NN * DH) / 4;
  const long long zeroA_n4 = ((long long)NN * DH) / 4;

  // ---- Phase 1: degrees + layer-1 aggregates ----
  zero_f4<<<(int)((zero1_n4 + 255) / 256), 256, 0, stream>>>((float4*)degI, zero1_n4);
  sage_scatter<<<scatterBlocks, 256, 0, stream>>>(x_user, e_u2i, e_u2i + nE,
                                                  agg1_item, degI, nE, DIN);
  sage_scatter<<<scatterBlocks, 256, 0, stream>>>(x_item, e_i2u, e_i2u + nE,
                                                  agg1_user, degU, nE, DIN);

  // ---- Layer-1 fused mean/GEMM/bias/ReLU (M=NN, N=256, K=128) ----
  const int g1Blocks = (NN / 16) * (DH / 64) / 8;     // 6250
  sage_gemm_wmma<<<g1Blocks, 256, 0, stream>>>(agg1_item, degI, x_item,
                                               Wl1_ui, Wr1_ui, b1_ui, hI,
                                               NN, DH, DIN, 1);
  sage_gemm_wmma<<<g1Blocks, 256, 0, stream>>>(agg1_user, degU, x_user,
                                               Wl1_iu, Wr1_iu, b1_iu, hU,
                                               NN, DH, DIN, 1);

  // ---- Layer 2 item: agg(h_user over u2i) -> o_item (N=128, K=256) ----
  const int g2Blocks = (NN / 16) * (DOUT / 64) / 8;   // 3125
  zero_f4<<<(int)((zeroA_n4 + 255) / 256), 256, 0, stream>>>((float4*)aggA, zeroA_n4);
  sage_scatter<<<scatterBlocks, 256, 0, stream>>>(hU, e_u2i, e_u2i + nE,
                                                  aggA, nullptr, nE, DH);
  sage_gemm_wmma<<<g2Blocks, 256, 0, stream>>>(aggA, degI, hI,
                                               Wl2_ui, Wr2_ui, b2_ui, o_item,
                                               NN, DOUT, DH, 0);

  // ---- Layer 2 user: agg(h_item over i2u) -> o_user ----
  zero_f4<<<(int)((zeroA_n4 + 255) / 256), 256, 0, stream>>>((float4*)aggA, zeroA_n4);
  sage_scatter<<<scatterBlocks, 256, 0, stream>>>(hI, e_i2u, e_i2u + nE,
                                                  aggA, nullptr, nE, DH);
  sage_gemm_wmma<<<g2Blocks, 256, 0, stream>>>(aggA, degU, hU,
                                               Wl2_iu, Wr2_iu, b2_iu, o_user,
                                               NN, DOUT, DH, 0);
}
